// WindowAttention_54726473286063
// MI455X (gfx1250) — matrix operand order repaired and checked by reading, not compiled, not run on hardware
//
#include <hip/hip_runtime.h>
#include <hip/hip_bf16.h>

typedef __attribute__((ext_vector_type(16))) __bf16 v16bf;
typedef __attribute__((ext_vector_type(8)))  float  v8f;

#define N_TOK 64
#define DIM 256
#define NUM_HEADS 8
#define HEAD_DIM 32
#define B_WIN 2048

// ---------------------------------------------------------------------------
// Kernel 1: meta-MLP relative-position bias.  bias[h][n][m], h<8, n,m<64.
// Batch-invariant, so compute once into workspace (8*64*64 floats = 128 KB).
// ---------------------------------------------------------------------------
__global__ void wa_bias_kernel(const float* __restrict__ rel,   // [64*64, 2]
                               const float* __restrict__ w1,    // [2, 256]
                               const float* __restrict__ b1,    // [256]
                               const float* __restrict__ w2,    // [256, 8]
                               const float* __restrict__ b2,    // [8]
                               float* __restrict__ bias)        // [8][64*64]
{
    int idx = blockIdx.x * blockDim.x + threadIdx.x;   // (n*64 + m)
    if (idx >= N_TOK * N_TOK) return;
    float x = rel[idx * 2 + 0];
    float y = rel[idx * 2 + 1];
    float acc[NUM_HEADS];
#pragma unroll
    for (int i = 0; i < NUM_HEADS; ++i) acc[i] = b2[i];
    for (int j = 0; j < 256; ++j) {
        float hj = fmaxf(x * w1[j] + y * w1[256 + j] + b1[j], 0.0f);
#pragma unroll
        for (int i = 0; i < NUM_HEADS; ++i) acc[i] += hj * w2[j * NUM_HEADS + i];
    }
#pragma unroll
    for (int i = 0; i < NUM_HEADS; ++i) bias[i * (N_TOK * N_TOK) + idx] = acc[i];
}

// ---------------------------------------------------------------------------
// Kernel 2: windowed attention, one block per (window, branch), one wave per
// head.  bf16 WMMA 16x16x32 for QK^T and PV; f32 softmax with per-head bias.
// ---------------------------------------------------------------------------
__global__ __launch_bounds__(256) void wa_attn_kernel(
    const float* __restrict__ qkv,      // [2048, 64, 768]
    const float* __restrict__ ass_qkv,  // [2048, 64, 768]
    const float* __restrict__ bias,     // [8][64][64]
    float* __restrict__ out)            // [2][2048, 64, 256] concat
{
    const int bid  = blockIdx.x;          // 0 .. 4095
    const int br   = bid >> 11;           // branch
    const int b    = bid & (B_WIN - 1);   // window
    const int tid  = threadIdx.x;
    const int wave = tid >> 5;            // == head index h
    const int lane = tid & 31;
    const int lh   = lane >> 4;           // lane half (0/1)
    const int ln   = lane & 15;
    const int h    = wave;

    const float* src = br ? ass_qkv : qkv;
    float* dst = out + (size_t)br * ((size_t)B_WIN * N_TOK * DIM);
    const float* base = src + (size_t)b * N_TOK * (3 * DIM);

    __shared__ float  Sbuf[NUM_HEADS][16][N_TOK];   // 32 KB: score row-block
    __shared__ __bf16 Pbuf[NUM_HEADS][16][N_TOK];   // 16 KB: softmaxed probs

    // ---- K as WMMA B operand (K^T: 32x16 per tile), 4 N-tiles ------------
    // B layout: lane -> column N=ln, K = lh*16 + 2v + e, pairs contiguous.
    v16bf kfrag[4];
#pragma unroll
    for (int nt = 0; nt < 4; ++nt) {
        const float* krow = base + (size_t)(nt * 16 + ln) * (3 * DIM)
                                 + DIM + h * HEAD_DIM + lh * 16;
        v16bf f;
#pragma unroll
        for (int v = 0; v < 8; ++v) {
            float2 p = *(const float2*)(krow + v * 2);
            f[2 * v]     = (__bf16)p.x;
            f[2 * v + 1] = (__bf16)p.y;
        }
        kfrag[nt] = f;
    }

    // ---- V as WMMA B operand (64x32), 2 K-chunks x 2 N-tiles -------------
    v16bf vfrag[2][2];
#pragma unroll
    for (int kt = 0; kt < 2; ++kt)
#pragma unroll
        for (int ntv = 0; ntv < 2; ++ntv) {
            v16bf f;
#pragma unroll
            for (int v = 0; v < 8; ++v) {
                int tok = kt * 32 + lh * 16 + v * 2;
                const float* p0 = base + (size_t)tok * (3 * DIM)
                                       + 2 * DIM + h * HEAD_DIM + ntv * 16 + ln;
                f[2 * v]     = (__bf16)p0[0];
                f[2 * v + 1] = (__bf16)p0[3 * DIM];   // tok+1
            }
            vfrag[kt][ntv] = f;
        }

    const float scale = 0.17677669529663687f;  // 1/sqrt(32)

    for (int mt = 0; mt < 4; ++mt) {
        // ---- Q A-fragment: contiguous layout k = lh*16 + v*2 ------
        v16bf qfrag;
        {
            const float* qrow = base + (size_t)(mt * 16 + ln) * (3 * DIM)
                                     + h * HEAD_DIM + lh * 16;
#pragma unroll
            for (int v = 0; v < 8; ++v) {
                int g = v >> 2, w = v & 3;
                float2 p = *(const float2*)(qrow + v * 2);
                qfrag[2 * v]     = (__bf16)(p.x * scale);
                qfrag[2 * v + 1] = (__bf16)(p.y * scale);
            }
        }

        // ---- scores: 4 WMMAs, spill to LDS via C/D layout ----------------
#pragma unroll
        for (int nt = 0; nt < 4; ++nt) {
            v8f acc = {};
            acc = __builtin_amdgcn_wmma_f32_16x16x32_bf16(
                      false, qfrag, false, kfrag[nt], (short)0, acc, false, false);
#pragma unroll
            for (int r = 0; r < 8; ++r) {
                int m = lh ? (r + 8) : r;   // VGPR r -> M=r (lanes 0-15), M=r+8
                Sbuf[wave][m][nt * 16 + ln] = acc[r];
            }
        }
        __syncthreads();

        // ---- softmax: lane owns row ln, half lh (32 cols) ----------------
        {
            float vals[32];
            const float* brow = bias + ((h * N_TOK + mt * 16 + ln) * N_TOK) + lh * 32;
            float mx = -3.0e38f;
#pragma unroll
            for (int i = 0; i < 32; ++i) {
                vals[i] = Sbuf[wave][ln][lh * 32 + i] + brow[i];
                mx = fmaxf(mx, vals[i]);
            }
            mx = fmaxf(mx, __shfl_xor(mx, 16, 32));
            float sum = 0.0f;
#pragma unroll
            for (int i = 0; i < 32; ++i) {
                vals[i] = __expf(vals[i] - mx);
                sum += vals[i];
            }
            sum += __shfl_xor(sum, 16, 32);
            float inv = __builtin_amdgcn_rcpf(sum);
#pragma unroll
            for (int i = 0; i < 32; ++i)
                Pbuf[wave][ln][lh * 32 + i] = (__bf16)(vals[i] * inv);
        }
        __syncthreads();

        // ---- P A-fragments from LDS (2 K-chunks) -------------------------
        v16bf pfrag[2];
#pragma unroll
        for (int kt = 0; kt < 2; ++kt) {
            v16bf f;
#pragma unroll
            for (int v = 0; v < 8; ++v) {
                int g = v >> 2, w = v & 3;
                int k = kt * 32 + lh * 16 + v * 2;
                f[2 * v]     = Pbuf[wave][ln][k];
                f[2 * v + 1] = Pbuf[wave][ln][k + 1];
            }
            pfrag[kt] = f;
        }

        // ---- O = P @ V : chained WMMAs, scatter per C/D layout -----------
#pragma unroll
        for (int ntv = 0; ntv < 2; ++ntv) {
            v8f o = {};
            o = __builtin_amdgcn_wmma_f32_16x16x32_bf16(
                    false, pfrag[0], false, vfrag[0][ntv], (short)0, o, false, false);
            o = __builtin_amdgcn_wmma_f32_16x16x32_bf16(
                    false, pfrag[1], false, vfrag[1][ntv], (short)0, o, false, false);
#pragma unroll
            for (int r = 0; r < 8; ++r) {
                int m = mt * 16 + (lh ? (r + 8) : r);
                dst[((size_t)b * N_TOK + m) * DIM + h * HEAD_DIM + ntv * 16 + ln] = o[r];
            }
        }
        __syncthreads();   // Sbuf/Pbuf reused next mt iteration
    }
}

// ---------------------------------------------------------------------------
extern "C" void kernel_launch(void* const* d_in, const int* in_sizes, int n_in,
                              void* d_out, int out_size, void* d_ws, size_t ws_size,
                              hipStream_t stream) {
    const float* qkv     = (const float*)d_in[0];
    const float* ass_qkv = (const float*)d_in[1];
    const float* rel_pos = (const float*)d_in[2];
    const float* w1      = (const float*)d_in[3];
    const float* b1      = (const float*)d_in[4];
    const float* w2      = (const float*)d_in[5];
    const float* b2      = (const float*)d_in[6];
    float* out  = (float*)d_out;
    float* bias = (float*)d_ws;   // 8*64*64 floats = 128 KB

    wa_bias_kernel<<<(N_TOK * N_TOK + 255) / 256, 256, 0, stream>>>(
        rel_pos, w1, b1, w2, b2, bias);

    wa_attn_kernel<<<B_WIN * 2, 256, 0, stream>>>(qkv, ass_qkv, bias, out);
}
